// DCNConv_74019466379803
// MI455X (gfx1250) — matrix operand order repaired and checked by reading, not compiled, not run on hardware
//
#include <hip/hip_runtime.h>
#include <hip/hip_bf16.h>
#include <stdint.h>

typedef __attribute__((ext_vector_type(16))) _Float16 v16h;
typedef __attribute__((ext_vector_type(8)))  float    v8f;
typedef __attribute__((ext_vector_type(4)))  unsigned int v4u;

// Problem constants (from reference)
#define NB   8
#define CC   256
#define HH   64
#define WW   64
#define LL   (HH * WW)          // 4096
#define GG   8
#define KK   9
#define CG   (CC / GG)          // 32
#define MM   (NB * LL)          // 32768 rows for all GEMMs
#define OO_OM (GG * KK * 3)     // 216
#define KDIM 256                // K for every projection GEMM

// ---------------- workspace layout (bytes) ----------------
#define OFF_XF16   ((size_t)0)                       // 32768*256 f16 = 16 MB
#define OFF_OMIN16 (OFF_XF16   + (size_t)MM * CC * 2)
#define OFF_VPW16  (OFF_OMIN16 + (size_t)MM * CC * 2)
#define OFF_OPW16  (OFF_VPW16  + (size_t)CC * CC * 2)
#define OFF_OMW16  (OFF_OPW16  + (size_t)CC * CC * 2)
#define OFF_VALUE  (OFF_OMW16  + (size_t)OO_OM * CC * 2)
#define OFF_OM     (OFF_VALUE  + (size_t)MM * CC * 4)
#define OFF_AGG16  (OFF_OM     + (size_t)MM * OO_OM * 4)

// ---------------- small conversion kernels ----------------
__global__ void cvt_f32_to_f16(const float* __restrict__ src,
                               _Float16* __restrict__ dst, int count) {
    int t = blockIdx.x * blockDim.x + threadIdx.x;
    if (t < count) dst[t] = (_Float16)src[t];
}

// x (N,C,H,W) f32 -> xf16 (N, L, C) f16   (A-matrix for value GEMM)
__global__ void xpose_kernel(const float* __restrict__ x,
                             _Float16* __restrict__ xf16) {
    size_t t = (size_t)blockIdx.x * blockDim.x + threadIdx.x;
    if (t >= (size_t)MM * CC) return;
    int c = (int)(t & (CC - 1));
    size_t ml = t >> 8;                 // n*L + l
    int l = (int)(ml & (LL - 1));
    int n = (int)(ml >> 12);
    xf16[t] = (_Float16)x[((size_t)n * CC + c) * LL + l];
}

// depthwise 3x3 SAME conv + bias, output as (N, L, C) f16 (A-matrix for om GEMM)
__global__ void dwconv_kernel(const float* __restrict__ x,
                              const float* __restrict__ dww,
                              const float* __restrict__ dwb,
                              _Float16* __restrict__ omin16) {
    size_t t = (size_t)blockIdx.x * blockDim.x + threadIdx.x;
    if (t >= (size_t)MM * CC) return;
    int c = (int)(t & (CC - 1));
    size_t ml = t >> 8;
    int l = (int)(ml & (LL - 1));
    int n = (int)(ml >> 12);
    int hh = l >> 6, ww = l & 63;
    const float* xp = x + ((size_t)n * CC + c) * LL;
    const float* wp = dww + c * 9;
    float s = dwb[c];
#pragma unroll
    for (int ky = 0; ky < 3; ++ky) {
        int y = hh + ky - 1;
        if ((unsigned)y >= (unsigned)HH) continue;
#pragma unroll
        for (int kx = 0; kx < 3; ++kx) {
            int xx = ww + kx - 1;
            if ((unsigned)xx >= (unsigned)WW) continue;
            s += wp[ky * 3 + kx] * xp[y * WW + xx];
        }
    }
    omin16[t] = (_Float16)s;
}

// ---------------- WMMA fragment helpers (contiguous k order, same in A and B) ----------------
// A: 16x32 f16, row m = lane&15; lanes 0-15 hold K 0-15, lanes 16-31 K 16-31.
__device__ inline v16h load_a_frag(const _Float16* __restrict__ A,
                                   int m0, int k0, int lane) {
    int ml = lane & 15, hi = lane >> 4;
    const _Float16* row = A + (size_t)(m0 + ml) * KDIM + k0;
    v16h a;
#pragma unroll
    for (int v = 0; v < 8; ++v) {
        int kb = 2 * v + hi * 16;
        a[2 * v]     = row[kb];
        a[2 * v + 1] = row[kb + 1];
    }
    return a;
}

// B fragment from the LDS-staged 64x256 block.
// Column o = o_local + lane&15; lanes 0-15 hold K 0-15, lanes 16-31 K 16-31.
__device__ inline v16h load_b_frag_lds(const _Float16* Bsm, int o_local,
                                       int k0, int lane) {
    int nl = lane & 15, hi = lane >> 4;
    const _Float16* row = Bsm + (size_t)(o_local + nl) * KDIM + k0 + hi * 16;
    v16h b;
#pragma unroll
    for (int v = 0; v < 8; ++v) {
        b[2 * v]     = row[2 * v];
        b[2 * v + 1] = row[2 * v + 1];
    }
    return b;
}

__device__ inline unsigned lds_off_u32(const void* p) {
    return (unsigned)(uintptr_t)(const __attribute__((address_space(3))) void*)p;
}

// ---------------- WMMA GEMM with async-LDS B staging ----------------
// Block = 8 waves: 8 M-tiles x one O-group (64 columns). B block (64x256 f16,
// 32 KB) staged once per block via global_load_async_to_lds_b128 (ASYNCcnt),
// then each wave runs 8 K-steps of 4 v_wmma off LDS-resident B.
// STORE_MODE 0: row-major f32 out[m*O + o]
// STORE_MODE 1: NCHW scatter   out[((n*CC + o)*LL) + l]
template <bool GUARD, int STORE_MODE>
__global__ void gemm_wmma_lds(const _Float16* __restrict__ A,
                              const _Float16* __restrict__ Wt,
                              const float* __restrict__ bias,
                              float* __restrict__ out,
                              int O, int Ogroups) {
    __shared__ __attribute__((aligned(16))) _Float16 Bsm[64 * KDIM]; // 32 KB
    int tid  = threadIdx.x;
    int lane = tid & 31;
    int wave = tid >> 5;
    int blockM = blockIdx.x / Ogroups;
    int og = blockIdx.x - blockM * Ogroups;
    int o0 = og << 6;

    if (GUARD) {
        // zero-fill so columns >= O contribute exactly 0
#pragma unroll
        for (int j = 0; j < 8; ++j) {
            int chunk = tid + j * 256;               // 16-byte chunks
            *(v4u*)(&Bsm[chunk * 8]) = (v4u){0u, 0u, 0u, 0u};
        }
        __syncthreads();
    }

    // async-stage B block: 2048 16-byte chunks over 256 threads
#pragma unroll
    for (int j = 0; j < 8; ++j) {
        int chunk = tid + j * 256;
        int orow = chunk >> 5;                       // 32 chunks per 512B row
        int within = chunk & 31;
        if (!GUARD || (o0 + orow) < O) {
            unsigned lds = lds_off_u32(&Bsm[chunk * 8]);
            unsigned long long ga = (unsigned long long)
                (Wt + (size_t)(o0 + orow) * KDIM + (size_t)within * 8);
            asm volatile("global_load_async_to_lds_b128 %0, %1, off"
                         :: "v"(lds), "v"(ga) : "memory");
        }
    }
    asm volatile("s_wait_asynccnt 0x0" ::: "memory");
    __syncthreads();

    int m0 = (blockM * 8 + wave) << 4;
    int ml = lane & 15;

    v8f acc0 = {}, acc1 = {}, acc2 = {}, acc3 = {};
#pragma unroll
    for (int k0 = 0; k0 < KDIM; k0 += 32) {
        if (k0 + 32 < KDIM)   // prefetch next K-slice of A row (global_prefetch_b8)
            __builtin_prefetch(A + (size_t)(m0 + ml) * KDIM + k0 + 32, 0, 1);
        v16h a  = load_a_frag(A, m0, k0, lane);
        v16h b0 = load_b_frag_lds(Bsm,  0, k0, lane);
        v16h b1 = load_b_frag_lds(Bsm, 16, k0, lane);
        v16h b2 = load_b_frag_lds(Bsm, 32, k0, lane);
        v16h b3 = load_b_frag_lds(Bsm, 48, k0, lane);
        acc0 = __builtin_amdgcn_wmma_f32_16x16x32_f16(false, a, false, b0,
                                                      (short)0, acc0, false, false);
        acc1 = __builtin_amdgcn_wmma_f32_16x16x32_f16(false, a, false, b1,
                                                      (short)0, acc1, false, false);
        acc2 = __builtin_amdgcn_wmma_f32_16x16x32_f16(false, a, false, b2,
                                                      (short)0, acc2, false, false);
        acc3 = __builtin_amdgcn_wmma_f32_16x16x32_f16(false, a, false, b3,
                                                      (short)0, acc3, false, false);
    }

    int nl = lane & 15, hi = lane >> 4;
#pragma unroll
    for (int sub = 0; sub < 4; ++sub) {
        v8f acc = (sub == 0) ? acc0 : (sub == 1) ? acc1 : (sub == 2) ? acc2 : acc3;
        int o = o0 + sub * 16 + nl;
        if (GUARD && o >= O) continue;
        float bv = bias ? bias[o] : 0.0f;
        if (STORE_MODE == 0) {
#pragma unroll
            for (int v = 0; v < 8; ++v) {
                int m = m0 + v + hi * 8;
                out[(size_t)m * O + o] = acc[v] + bv;
            }
        } else {
#pragma unroll
            for (int v = 0; v < 8; ++v) {
                int m = m0 + v + hi * 8;
                int n = m >> 12;             // L = 4096
                int l = m & (LL - 1);
                out[((size_t)n * CC + o) * LL + l] = acc[v] + bv;
            }
        }
    }
}

// ---------------- bilinear deformable sampling + mask aggregation ----------------
// one thread per (n, l, g); accumulates CG=32 channels; writes agg (N,L,C) f16
__global__ void sample_kernel(const float* __restrict__ value,   // (M, C) f32
                              const float* __restrict__ om,      // (M, 216) f32
                              _Float16* __restrict__ agg16) {    // (M, C) f16
    int t = blockIdx.x * blockDim.x + threadIdx.x;
    if (t >= MM * GG) return;
    int g = t & (GG - 1);
    int m = t >> 3;
    int l = m & (LL - 1);
    int n = m >> 12;
    int hh = l >> 6, ww = l & 63;

    const float* omr = om + (size_t)m * OO_OM + g * (KK * 3);
    const float* vb  = value + (size_t)n * LL * CC + g * CG;

    float acc[CG];
#pragma unroll
    for (int c = 0; c < CG; ++c) acc[c] = 0.0f;

#pragma unroll
    for (int k = 0; k < KK; ++k) {
        float dy = (float)(k / 3 - 1);
        float dx = (float)(k % 3 - 1);
        float py = (float)hh + dy + omr[2 * k];       // OFFSET_SCALE = 1
        float px = (float)ww + dx + omr[2 * k + 1];
        float msk = omr[2 * KK + k];
        float y0f = floorf(py), x0f = floorf(px);
        float ty = py - y0f, tx = px - x0f;
        int y0 = (int)y0f, x0 = (int)x0f;
        float wc[4] = { (1.f - ty) * (1.f - tx), (1.f - ty) * tx,
                        ty * (1.f - tx),          ty * tx };
        int yy[4] = { y0, y0, y0 + 1, y0 + 1 };
        int xx[4] = { x0, x0 + 1, x0, x0 + 1 };
#pragma unroll
        for (int q = 0; q < 4; ++q) {
            bool valid = (yy[q] >= 0) & (yy[q] < HH) & (xx[q] >= 0) & (xx[q] < WW);
            float wgt = msk * wc[q] * (valid ? 1.0f : 0.0f);
            int yi = min(max(yy[q], 0), HH - 1);
            int xi = min(max(xx[q], 0), WW - 1);
            const float* vr = vb + (size_t)(yi * WW + xi) * CC;
#pragma unroll
            for (int c = 0; c < CG; ++c) acc[c] += wgt * vr[c];
        }
    }

    _Float16* dst = agg16 + (size_t)m * CC + g * CG;
#pragma unroll
    for (int c = 0; c < CG; ++c) dst[c] = (_Float16)acc[c];
}

// ---------------- host launcher ----------------
extern "C" void kernel_launch(void* const* d_in, const int* in_sizes, int n_in,
                              void* d_out, int out_size, void* d_ws, size_t ws_size,
                              hipStream_t stream) {
    const float* x    = (const float*)d_in[0];
    const float* dw_w = (const float*)d_in[1];
    const float* dw_b = (const float*)d_in[2];
    const float* om_w = (const float*)d_in[3];
    const float* om_b = (const float*)d_in[4];
    const float* vp_w = (const float*)d_in[5];
    const float* vp_b = (const float*)d_in[6];
    const float* op_w = (const float*)d_in[7];
    float* out = (float*)d_out;

    char* ws = (char*)d_ws;
    _Float16* xf16   = (_Float16*)(ws + OFF_XF16);
    _Float16* omin16 = (_Float16*)(ws + OFF_OMIN16);
    _Float16* vpw16  = (_Float16*)(ws + OFF_VPW16);
    _Float16* opw16  = (_Float16*)(ws + OFF_OPW16);
    _Float16* omw16  = (_Float16*)(ws + OFF_OMW16);
    float*    value  = (float*)(ws + OFF_VALUE);
    float*    omb    = (float*)(ws + OFF_OM);
    _Float16* agg16  = (_Float16*)(ws + OFF_AGG16);

    const int T = 256;
    const int Mtiles = MM / 16;          // 2048, divisible by 8 waves/block

    // 1) weight conversions
    cvt_f32_to_f16<<<(CC * CC + T - 1) / T, T, 0, stream>>>(vp_w, vpw16, CC * CC);
    cvt_f32_to_f16<<<(CC * CC + T - 1) / T, T, 0, stream>>>(op_w, opw16, CC * CC);
    cvt_f32_to_f16<<<(OO_OM * CC + T - 1) / T, T, 0, stream>>>(om_w, omw16, OO_OM * CC);

    // 2) x transpose -> (N,L,C) f16
    {
        size_t tot = (size_t)MM * CC;
        xpose_kernel<<<(unsigned)((tot + T - 1) / T), T, 0, stream>>>(x, xf16);
    }

    // 3) depthwise conv -> (N,L,C) f16
    {
        size_t tot = (size_t)MM * CC;
        dwconv_kernel<<<(unsigned)((tot + T - 1) / T), T, 0, stream>>>(x, dw_w, dw_b, omin16);
    }

    // 4) value = xf16 @ vp_w^T + vp_b   (O = 256, unguarded)
    {
        int Ogroups = CC / 64;
        int blocks = (Mtiles / 8) * Ogroups;
        gemm_wmma_lds<false, 0><<<blocks, T, 0, stream>>>(
            xf16, vpw16, vp_b, value, CC, Ogroups);
    }

    // 5) om = omin16 @ om_w^T + om_b    (O = 216, guarded / zero-padded LDS rows)
    {
        int Ogroups = (OO_OM + 63) / 64;
        int blocks = (Mtiles / 8) * Ogroups;
        gemm_wmma_lds<true, 0><<<blocks, T, 0, stream>>>(
            omin16, omw16, om_b, omb, OO_OM, Ogroups);
    }

    // 6) deformable sampling + aggregation -> agg f16
    {
        int tot = MM * GG;
        sample_kernel<<<(tot + T - 1) / T, T, 0, stream>>>(value, omb, agg16);
    }

    // 7) out = agg @ op_w^T, scattered to NCHW
    {
        int Ogroups = CC / 64;
        int blocks = (Mtiles / 8) * Ogroups;
        gemm_wmma_lds<false, 1><<<blocks, T, 0, stream>>>(
            agg16, opw16, nullptr, out, CC, Ogroups);
    }
}
